// DeBERTaMoEClassifier_25374666784925
// MI455X (gfx1250) — matrix operand order repaired and checked by reading, not compiled, not run on hardware
//
#include <hip/hip_runtime.h>
#include <hip/hip_bf16.h>
#include <math.h>

// Problem dims (fixed by the reference)
#define BB 1024
#define SS 256
#define HH 1024
#define EE 16
#define KTOP 4
#define HE 256
#define CC 3

typedef __bf16 bf16_t;
typedef __attribute__((ext_vector_type(16))) __bf16 v16bf;
typedef __attribute__((ext_vector_type(8)))  float  v8f;
typedef __attribute__((ext_vector_type(4)))  float  f32x4;

#if __has_builtin(__builtin_amdgcn_tensor_load_to_lds) && __has_builtin(__builtin_amdgcn_s_wait_tensorcnt)
#define USE_TDM 1
#else
#define USE_TDM 0
#endif

__device__ __forceinline__ v8f wmma_bf16(v16bf a, v16bf b, v8f c) {
    // D = A(16x32 bf16) * B(32x16 bf16) + C(16x16 f32)
    return __builtin_amdgcn_wmma_f32_16x16x32_bf16(
        /*neg_a=*/false, a, /*neg_b=*/false, b,
        /*c_mod=*/(short)0, c, /*reuse_a=*/false, /*reuse_b=*/false);
}

#if USE_TDM
typedef __attribute__((ext_vector_type(4))) unsigned int u32x4;
typedef __attribute__((ext_vector_type(8))) int         i32x8;
typedef __attribute__((ext_vector_type(4))) int         i32x4;

// One TDM DMA: copy (elems0 x rows) bf16 elements, contiguous in global
// (row stride == elems0), into LDS at `l`. D# per ISA 08_async_tensor §8.3/8.4.
// elems0 must fit tile_dim0 (16 bits).
__device__ __forceinline__ void tdm_load_bf16(const bf16_t* g, bf16_t* l,
                                              unsigned elems0, unsigned rows) {
    unsigned long long ga = (unsigned long long)g;
    unsigned lds = (unsigned)(unsigned long long)l;   // low 32 bits = LDS offset
    u32x4 g0;
    g0[0] = 1u;                                        // count=1, user descriptor
    g0[1] = lds;                                       // lds_addr
    g0[2] = (unsigned)(ga & 0xffffffffu);              // global_addr[31:0]
    g0[3] = (unsigned)((ga >> 32) & 0x01ffffffu)       // global_addr[56:32]
            | (2u << 30);                              // type = 2 ("image")
    i32x8 g1;
    g1[0] = (int)(1u << 16);                           // data_size=1 (2 bytes), wg_mask=0
    g1[1] = (int)((elems0 & 0xffffu) << 16);           // tensor_dim0[15:0] @ [63:48]
    g1[2] = (int)((elems0 >> 16) | (rows << 16));      // tensor_dim0[31:16], tensor_dim1[15:0]
    g1[3] = (int)((elems0 & 0xffffu) << 16);           // tile_dim0 @ [127:112]
    g1[4] = (int)rows;                                 // tile_dim1 (tile_dim2=0)
    g1[5] = (int)elems0;                               // tensor_dim0_stride[31:0]
    g1[6] = 0;                                         // stride0[47:32]=0, stride1 lo=0
    g1[7] = 0;                                         // stride1 hi=0
    i32x4 gz4 = {0, 0, 0, 0};                          // groups 2/3 unused (<=2D)
    i32x8 gz8 = {0, 0, 0, 0, 0, 0, 0, 0};              // extra group (6-arg form)
    __builtin_amdgcn_tensor_load_to_lds(g0, g1, gz4, gz4, gz8, 0);
}
#endif

// ---------------------------------------------------------------------------
// 1) pooled[b][h] = mean_s hidden[b][s][h]   (the 1 GB streaming pass)
//    Non-temporal loads: this stream is read-once, keep it out of L2 so the
//    GEMM weight working set (~14 MB) stays resident.
// ---------------------------------------------------------------------------
__global__ void pool_kernel(const float* __restrict__ hs, float* __restrict__ pooled) {
    int b = blockIdx.x;
    int t = threadIdx.x;                       // 256 threads, 4 cols each
    const float* base = hs + (size_t)b * SS * HH + (size_t)t * 4;
    float ax = 0.f, ay = 0.f, az = 0.f, aw = 0.f;
#pragma unroll 4
    for (int s = 0; s < SS; ++s) {
        f32x4 v = __builtin_nontemporal_load((const f32x4*)(base + (size_t)s * HH));
        ax += v.x; ay += v.y; az += v.z; aw += v.w;
    }
    const float inv = 1.0f / (float)SS;
    f32x4 r; r.x = ax * inv; r.y = ay * inv; r.z = az * inv; r.w = aw * inv;
    *(f32x4*)(pooled + (size_t)b * HH + (size_t)t * 4) = r;
}

// ---------------------------------------------------------------------------
// 2) f32 -> bf16 repack into WMMA operand layouts (ISA 7.12.2).
//    Tile = 512 contiguous bf16 laid out as [lane(32)][halfword(16)].
// ---------------------------------------------------------------------------
__global__ void packA_kernel(const float* __restrict__ X, int M, int Kd,
                             long rowStride, bf16_t* __restrict__ dst) {
    int kTiles = Kd >> 5;
    size_t total = (size_t)M * Kd;
    size_t stride = (size_t)gridDim.x * blockDim.x;
    for (size_t o = (size_t)blockIdx.x * blockDim.x + threadIdx.x; o < total; o += stride) {
        size_t tile = o >> 9; int within = (int)(o & 511);
        int lane = within >> 4, i = within & 15;
        int mT = (int)(tile / kTiles), kT = (int)(tile % kTiles);
        int m = mT * 16 + (lane & 15);
        int half = lane >> 4;
        int v = i >> 1, p = i & 1;
        int k = kT * 32 + (v >> 2) * 16 + half * 8 + (v & 3) * 2 + p;
        dst[o] = (bf16_t)X[(size_t)m * rowStride + k];
    }
}

// B operand (32x16, 16-bit): n = lane&15; k = (i>>3)*16 + (lane>>4)*8 + (i&7). Batched over E.
__global__ void packB_kernel(const float* __restrict__ W, int N, int Kd, int batch,
                             bf16_t* __restrict__ dst) {
    int kTiles = Kd >> 5;
    size_t per = (size_t)N * Kd;
    size_t total = per * batch;
    size_t stride = (size_t)gridDim.x * blockDim.x;
    for (size_t o = (size_t)blockIdx.x * blockDim.x + threadIdx.x; o < total; o += stride) {
        int e = (int)(o / per); size_t r = o % per;
        size_t tile = r >> 9; int within = (int)(r & 511);
        int lane = within >> 4, i = within & 15;
        int nT = (int)(tile / kTiles), kT = (int)(tile % kTiles);
        int n = nT * 16 + (lane & 15);
        int k = kT * 32 + (i >> 3) * 16 + (lane >> 4) * 8 + (i & 7);
        dst[o] = (bf16_t)W[(size_t)e * per + (size_t)n * Kd + k];
    }
}

// ---------------------------------------------------------------------------
// 3) Router: logits = cls @ router_W.T + b ; top-4 ; softmax
// ---------------------------------------------------------------------------
__global__ void router_kernel(const float* __restrict__ hs,
                              const float* __restrict__ rW, const float* __restrict__ rb,
                              int* __restrict__ topi, float* __restrict__ topw) {
    int b = blockIdx.x;
    int tid = threadIdx.x;                 // 256 = 16 experts x 16 threads
    int e = tid >> 4, l = tid & 15;
    const float* cls = hs + (size_t)b * SS * HH;  // row s == 0
    float sum = 0.f;
    for (int k = l; k < HH; k += 16) sum += cls[k] * rW[e * HH + k];
#pragma unroll
    for (int off = 1; off < 16; off <<= 1) sum += __shfl_xor(sum, off, 32);
    __shared__ float lg[EE];
    if (l == 0) lg[e] = sum + rb[e];
    __syncthreads();
    if (tid == 0) {
        float v[EE];
#pragma unroll
        for (int i = 0; i < EE; ++i) v[i] = lg[i];
        int idx[KTOP]; float val[KTOP];
        for (int kk = 0; kk < KTOP; ++kk) {
            int best = 0; float bv = -1e30f;
            for (int i = 0; i < EE; ++i) if (v[i] > bv) { bv = v[i]; best = i; }
            idx[kk] = best; val[kk] = bv; v[best] = -1e30f;
        }
        float mx = val[0], s = 0.f, w[KTOP];
        for (int kk = 0; kk < KTOP; ++kk) { w[kk] = __expf(val[kk] - mx); s += w[kk]; }
        for (int kk = 0; kk < KTOP; ++kk) {
            topi[b * KTOP + kk] = idx[kk];
            topw[b * KTOP + kk] = w[kk] / s;
        }
    }
}

// ---------------------------------------------------------------------------
// 4) orig = tanh(cls @ dense_W.T + dense_b) @ out_W.T + out_b
//    Block = 16 tokens x full 1024 cols; 8 waves x 8 n-tiles; K=1024.
//    A tile (32 KB) staged once per block via TDM; B streamed from L2.
// ---------------------------------------------------------------------------
__global__ void dense_kernel(const bf16_t* __restrict__ Acls, const bf16_t* __restrict__ Bd,
                             const float* __restrict__ dense_b, const float* __restrict__ out_W,
                             const float* __restrict__ out_b, float* __restrict__ orig) {
    const int KT = HH / 32;                     // 32 k-tiles
    int mTile = blockIdx.x;
    int wave = threadIdx.x >> 5, lane = threadIdx.x & 31;
    __shared__ float sacc[16][CC];
#if USE_TDM
    __shared__ __attribute__((aligned(32))) bf16_t aTile[32 * 512];   // 32 KB
    if (wave == 0)
        tdm_load_bf16(Acls + ((size_t)mTile * KT) * 512, aTile, KT * 512, 1);
#endif
    if (threadIdx.x < 16 * CC) sacc[threadIdx.x / CC][threadIdx.x % CC] = 0.f;
#if USE_TDM
    if (wave == 0) __builtin_amdgcn_s_wait_tensorcnt(0);
#endif
    __syncthreads();

    v8f zero = {};
    v8f acc[8];
#pragma unroll
    for (int nt = 0; nt < 8; ++nt) acc[nt] = zero;

#if !USE_TDM
    const bf16_t* Ab = Acls + ((size_t)mTile * KT) * 512 + lane * 16;
#endif
    for (int kt = 0; kt < KT; ++kt) {
#if USE_TDM
        v16bf a = *(const v16bf*)(aTile + kt * 512 + lane * 16);
#else
        v16bf a = *(const v16bf*)(Ab + (size_t)kt * 512);
#endif
#pragma unroll
        for (int nt = 0; nt < 8; ++nt) {
            int nTile = wave * 8 + nt;
            v16bf bm = *(const v16bf*)(Bd + ((size_t)nTile * KT + kt) * 512 + lane * 16);
            acc[nt] = wmma_bf16(a, bm, acc[nt]);
        }
    }

    // Fused epilogue: tanh, then contract against out_W (3 rows) per lane.
    int col = lane & 15, rg = lane >> 4;
    float pc[8][CC];
#pragma unroll
    for (int r = 0; r < 8; ++r)
#pragma unroll
        for (int c = 0; c < CC; ++c) pc[r][c] = 0.f;

#pragma unroll
    for (int nt = 0; nt < 8; ++nt) {
        int n = (wave * 8 + nt) * 16 + col;
        float bn = dense_b[n];
        float w0 = out_W[0 * HH + n], w1 = out_W[1 * HH + n], w2 = out_W[2 * HH + n];
#pragma unroll
        for (int r = 0; r < 8; ++r) {
            float t = tanhf(acc[nt][r] + bn);
            pc[r][0] += t * w0; pc[r][1] += t * w1; pc[r][2] += t * w2;
        }
    }
#pragma unroll
    for (int off = 1; off < 16; off <<= 1)
#pragma unroll
        for (int r = 0; r < 8; ++r)
#pragma unroll
            for (int c = 0; c < CC; ++c) pc[r][c] += __shfl_xor(pc[r][c], off, 32);
    if ((lane & 15) == 0)
#pragma unroll
        for (int r = 0; r < 8; ++r)
#pragma unroll
            for (int c = 0; c < CC; ++c) atomicAdd(&sacc[rg * 8 + r][c], pc[r][c]);
    __syncthreads();
    if (threadIdx.x < 16 * CC) {
        int m = threadIdx.x / CC, c = threadIdx.x % CC;
        orig[(size_t)(mTile * 16 + m) * CC + c] = sacc[m][c] + out_b[c];
    }
}

// ---------------------------------------------------------------------------
// 5) Expert MLP per (m-tile, expert): GEMM1 -> LN -> GELU -> GEMM2 -> proj(3)
//    Pooled A tile staged via TDM and shared by all 8 waves.
// ---------------------------------------------------------------------------
__global__ void expert_kernel(const bf16_t* __restrict__ Apool,
                              const bf16_t* __restrict__ W1p, const bf16_t* __restrict__ W2p,
                              const float* __restrict__ eb1, const float* __restrict__ eg,
                              const float* __restrict__ ebt, const float* __restrict__ eb2,
                              const float* __restrict__ proj_W, const float* __restrict__ proj_b,
                              float* __restrict__ el) {
    const int KT1 = HH / 32;                    // 32
    const int KT2 = HE / 32;                    // 8
    int mTile = blockIdx.x, e = blockIdx.y;
    int wave = threadIdx.x >> 5, lane = threadIdx.x & 31;
    int col = lane & 15, rg = lane >> 4;

    __shared__ float hs1[16 * HE];                                  // 16 KB h1 tile
    __shared__ __attribute__((aligned(32))) bf16_t ap2[8 * 512];    // 8 KB packed A2
    __shared__ float sacc[16][CC];
    __shared__ float smean[16], srstd[16];
#if USE_TDM
    __shared__ __attribute__((aligned(32))) bf16_t aTile[32 * 512]; // 32 KB packed A1
    if (wave == 0) {
        tdm_load_bf16(Apool + ((size_t)mTile * KT1) * 512, aTile, KT1 * 512, 1);
        __builtin_amdgcn_s_wait_tensorcnt(0);
    }
    __syncthreads();
#endif

    v8f zero = {};
    v8f acc[2]; acc[0] = zero; acc[1] = zero;

    // ---- GEMM1: h1 = pooled @ eW1[e].T  (M=16, N=256, K=1024)
#if !USE_TDM
    const bf16_t* Ab = Apool + ((size_t)mTile * KT1) * 512 + lane * 16;
#endif
    const bf16_t* B1 = W1p + (size_t)e * (HE / 16) * KT1 * 512;
    for (int kt = 0; kt < KT1; ++kt) {
#if USE_TDM
        v16bf a = *(const v16bf*)(aTile + kt * 512 + lane * 16);
#else
        v16bf a = *(const v16bf*)(Ab + (size_t)kt * 512);
#endif
#pragma unroll
        for (int nt = 0; nt < 2; ++nt) {
            int nTile = wave * 2 + nt;
            v16bf bm = *(const v16bf*)(B1 + ((size_t)nTile * KT1 + kt) * 512 + lane * 16);
            acc[nt] = wmma_bf16(a, bm, acc[nt]);
        }
    }
    // Spill h1 (+bias) to LDS in (m, n) order.
#pragma unroll
    for (int nt = 0; nt < 2; ++nt) {
        int n = (wave * 2 + nt) * 16 + col;
        float bn = eb1[e * HE + n];
#pragma unroll
        for (int r = 0; r < 8; ++r) hs1[(rg * 8 + r) * HE + n] = acc[nt][r] + bn;
    }
    if (threadIdx.x < 16 * CC) sacc[threadIdx.x / CC][threadIdx.x % CC] = 0.f;
    __syncthreads();

    // ---- LayerNorm stats: 16 threads per row, 16 cols each
    {
        int row = threadIdx.x >> 4, c0 = (threadIdx.x & 15) * 16;
        float s1 = 0.f, s2 = 0.f;
#pragma unroll
        for (int j = 0; j < 16; ++j) { float x = hs1[row * HE + c0 + j]; s1 += x; s2 += x * x; }
#pragma unroll
        for (int off = 1; off < 16; off <<= 1) { s1 += __shfl_xor(s1, off, 32); s2 += __shfl_xor(s2, off, 32); }
        if ((threadIdx.x & 15) == 0) {
            float m = s1 / (float)HE;
            float var = s2 / (float)HE - m * m;
            smean[row] = m; srstd[row] = rsqrtf(var + 1e-5f);
        }
    }
    __syncthreads();

    // ---- normalize + exact GELU, write straight into WMMA-A layout (bf16)
    {
        int row = threadIdx.x >> 4, c0 = (threadIdx.x & 15) * 16;
        float m = smean[row], rs = srstd[row];
#pragma unroll
        for (int j = 0; j < 16; ++j) {
            int k = c0 + j;
            float x = hs1[row * HE + k];
            float y = (x - m) * rs * eg[e * HE + k] + ebt[e * HE + k];
            float g = 0.5f * y * (1.0f + erff(y * 0.70710678118654752f));
            int kT = k >> 5, kk = k & 31;
            int grp = kk >> 4, rem = kk & 15, half = rem >> 3;
            int v = grp * 4 + ((rem & 7) >> 1), p = rem & 1;
            ap2[kT * 512 + (half * 16 + row) * 16 + v * 2 + p] = (bf16_t)g;
        }
    }
    __syncthreads();

    // ---- GEMM2: h2 = g @ eW2[e].T  (M=16, N=256, K=256), A from LDS
    v8f acc2[2]; acc2[0] = zero; acc2[1] = zero;
    const bf16_t* B2 = W2p + (size_t)e * (HE / 16) * KT2 * 512;
    for (int kt = 0; kt < KT2; ++kt) {
        v16bf a = *(const v16bf*)(ap2 + kt * 512 + lane * 16);
#pragma unroll
        for (int nt = 0; nt < 2; ++nt) {
            int nTile = wave * 2 + nt;
            v16bf bm = *(const v16bf*)(B2 + ((size_t)nTile * KT2 + kt) * 512 + lane * 16);
            acc2[nt] = wmma_bf16(a, bm, acc2[nt]);
        }
    }

    // ---- fused projection to C=3 expert logits
    float pc[8][CC];
#pragma unroll
    for (int r = 0; r < 8; ++r)
#pragma unroll
        for (int c = 0; c < CC; ++c) pc[r][c] = 0.f;
#pragma unroll
    for (int nt = 0; nt < 2; ++nt) {
        int n = (wave * 2 + nt) * 16 + col;
        float bn = eb2[e * HE + n];
        float w0 = proj_W[0 * HE + n], w1 = proj_W[1 * HE + n], w2 = proj_W[2 * HE + n];
#pragma unroll
        for (int r = 0; r < 8; ++r) {
            float h2 = acc2[nt][r] + bn;
            pc[r][0] += h2 * w0; pc[r][1] += h2 * w1; pc[r][2] += h2 * w2;
        }
    }
#pragma unroll
    for (int off = 1; off < 16; off <<= 1)
#pragma unroll
        for (int r = 0; r < 8; ++r)
#pragma unroll
            for (int c = 0; c < CC; ++c) pc[r][c] += __shfl_xor(pc[r][c], off, 32);
    if ((lane & 15) == 0)
#pragma unroll
        for (int r = 0; r < 8; ++r)
#pragma unroll
            for (int c = 0; c < CC; ++c) atomicAdd(&sacc[rg * 8 + r][c], pc[r][c]);
    __syncthreads();
    if (threadIdx.x < 16 * CC) {
        int m = threadIdx.x / CC, c = threadIdx.x % CC;
        int b = mTile * 16 + m;
        el[((size_t)e * BB + b) * CC + c] = sacc[m][c] + proj_b[c];
    }
}

// ---------------------------------------------------------------------------
// 6) Final head: top-4 mix, concat, 6->3 MLP with LN + ReLU (one thread/token)
// ---------------------------------------------------------------------------
__global__ void final_kernel(const float* __restrict__ orig, const float* __restrict__ el,
                             const int* __restrict__ topi, const float* __restrict__ topw,
                             const float* __restrict__ f1_W, const float* __restrict__ f1_b,
                             const float* __restrict__ fg, const float* __restrict__ fbt,
                             const float* __restrict__ f2_W, const float* __restrict__ f2_b,
                             float* __restrict__ out) {
    int b = blockIdx.x * blockDim.x + threadIdx.x;
    if (b >= BB) return;
    float comb[2 * CC];
#pragma unroll
    for (int c = 0; c < CC; ++c) comb[c] = orig[(size_t)b * CC + c];
    float m0 = 0.f, m1 = 0.f, m2 = 0.f;
#pragma unroll
    for (int k = 0; k < KTOP; ++k) {
        int e = topi[b * KTOP + k]; float w = topw[b * KTOP + k];
        const float* p = el + ((size_t)e * BB + b) * CC;
        m0 += w * p[0]; m1 += w * p[1]; m2 += w * p[2];
    }
    comb[3] = m0; comb[4] = m1; comb[5] = m2;
    float y[CC];
#pragma unroll
    for (int c = 0; c < CC; ++c) {
        float s = f1_b[c];
#pragma unroll
        for (int j = 0; j < 2 * CC; ++j) s += comb[j] * f1_W[c * 2 * CC + j];
        y[c] = s;
    }
    float mean = (y[0] + y[1] + y[2]) * (1.f / 3.f);
    float var = ((y[0] - mean) * (y[0] - mean) + (y[1] - mean) * (y[1] - mean) +
                 (y[2] - mean) * (y[2] - mean)) * (1.f / 3.f);
    float rs = rsqrtf(var + 1e-5f);
    float z[CC];
#pragma unroll
    for (int c = 0; c < CC; ++c) {
        float t = (y[c] - mean) * rs * fg[c] + fbt[c];
        z[c] = t > 0.f ? t : 0.f;
    }
#pragma unroll
    for (int c = 0; c < CC; ++c) {
        float s = f2_b[c];
#pragma unroll
        for (int j = 0; j < CC; ++j) s += z[j] * f2_W[c * CC + j];
        out[(size_t)b * CC + c] = s;
    }
}

// ---------------------------------------------------------------------------
extern "C" void kernel_launch(void* const* d_in, const int* in_sizes, int n_in,
                              void* d_out, int out_size, void* d_ws, size_t ws_size,
                              hipStream_t stream) {
    const float* hidden   = (const float*)d_in[0];
    const float* router_W = (const float*)d_in[1];
    const float* router_b = (const float*)d_in[2];
    const float* eW1      = (const float*)d_in[3];
    const float* eb1      = (const float*)d_in[4];
    const float* eg       = (const float*)d_in[5];
    const float* ebt      = (const float*)d_in[6];
    const float* eW2      = (const float*)d_in[7];
    const float* eb2      = (const float*)d_in[8];
    const float* proj_W   = (const float*)d_in[9];
    const float* proj_b   = (const float*)d_in[10];
    const float* dense_W  = (const float*)d_in[11];
    const float* dense_b  = (const float*)d_in[12];
    const float* out_W    = (const float*)d_in[13];
    const float* out_b    = (const float*)d_in[14];
    const float* f1_W     = (const float*)d_in[15];
    const float* f1_b     = (const float*)d_in[16];
    const float* fg       = (const float*)d_in[17];
    const float* fbt      = (const float*)d_in[18];
    const float* f2_W     = (const float*)d_in[19];
    const float* f2_b     = (const float*)d_in[20];

    // Workspace carve-out (256B aligned slices)
    size_t off = 0;
    auto carve = [&](size_t bytes) -> void* {
        void* p = (void*)((char*)d_ws + off);
        off += (bytes + 255) & ~(size_t)255;
        return p;
    };
    float*  pooled = (float*)carve((size_t)BB * HH * sizeof(float));
    bf16_t* Acls   = (bf16_t*)carve((size_t)BB * HH * sizeof(bf16_t));
    bf16_t* Apool  = (bf16_t*)carve((size_t)BB * HH * sizeof(bf16_t));
    bf16_t* Bdense = (bf16_t*)carve((size_t)HH * HH * sizeof(bf16_t));
    bf16_t* W1p    = (bf16_t*)carve((size_t)EE * HE * HH * sizeof(bf16_t));
    bf16_t* W2p    = (bf16_t*)carve((size_t)EE * HE * HE * sizeof(bf16_t));
    float*  orig   = (float*)carve((size_t)BB * CC * sizeof(float));
    float*  el     = (float*)carve((size_t)EE * BB * CC * sizeof(float));
    int*    topi   = (int*)carve((size_t)BB * KTOP * sizeof(int));
    float*  topw   = (float*)carve((size_t)BB * KTOP * sizeof(float));
    (void)ws_size; (void)in_sizes; (void)n_in; (void)out_size;

    // 1) pooling (dominant 1 GB HBM pass, non-temporal)
    pool_kernel<<<BB, 256, 0, stream>>>(hidden, pooled);

    // 2) bf16 repacks into WMMA operand layouts
    packA_kernel<<<(BB * HH) / 256, 256, 0, stream>>>(hidden, BB, HH, (long)SS * HH, Acls);
    packA_kernel<<<(BB * HH) / 256, 256, 0, stream>>>(pooled, BB, HH, (long)HH, Apool);
    packB_kernel<<<(HH * HH) / 256, 256, 0, stream>>>(dense_W, HH, HH, 1, Bdense);
    packB_kernel<<<(EE * HE * HH) / 256, 256, 0, stream>>>(eW1, HE, HH, EE, W1p);
    packB_kernel<<<(EE * HE * HE) / 256, 256, 0, stream>>>(eW2, HE, HE, EE, W2p);

    // 3) router top-k
    router_kernel<<<BB, 256, 0, stream>>>(hidden, router_W, router_b, topi, topw);

    // 4) dense path (TDM A-stage + WMMA + fused tanh + out_W contraction)
    dense_kernel<<<BB / 16, 256, 0, stream>>>(Acls, Bdense, dense_b, out_W, out_b, orig);

    // 5) expert MLPs (TDM A-stage + WMMA x2, fused LN/GELU/proj)
    expert_kernel<<<dim3(BB / 16, EE), 256, 0, stream>>>(Apool, W1p, W2p, eb1, eg, ebt,
                                                         eb2, proj_W, proj_b, el);

    // 6) final head
    final_kernel<<<BB / 256, 256, 0, stream>>>(orig, el, topi, topw, f1_W, f1_b, fg, fbt,
                                               f2_W, f2_b, (float*)d_out);
}
